// MultiHeadSelfAttention_6588479832622
// MI455X (gfx1250) — matrix operand order repaired and checked by reading, not compiled, not run on hardware
//
#include <hip/hip_runtime.h>
#include <stdint.h>

// ---------------------------------------------------------------------------
// Swin-window MHSA for MI455X (gfx1250, wave32):
//   - all GEMMs via v_wmma_f32_16x16x32_bf16 (fp32 accumulate)
//   - GEMM tile staging via Tensor Data Mover (tensor_load_to_lds, TENSORcnt),
//     double-buffered in LDS; cooperative-load fallback if builtin is absent
// ---------------------------------------------------------------------------
#define D_EMB  768
#define N_HEADS 12
#define SEQLEN 196
#define PD      64
#define NBATCH  64
#define MROWS  (NBATCH * SEQLEN)   // 12544 = 98 * 128
#define NQKV   (3 * D_EMB)         // 2304  = 18 * 128
#define SPAD   224                 // key dim padded: 14*16 = 7*32
#define QPAD   208                 // query rows padded: 13*16

typedef __attribute__((ext_vector_type(16))) __bf16 bf16x16;
typedef __attribute__((ext_vector_type(8)))  float  f32x8;
typedef __attribute__((ext_vector_type(4)))  unsigned u32x4;
typedef __attribute__((ext_vector_type(8)))  int      i32x8;
typedef __attribute__((ext_vector_type(4)))  int      i32x4;

#if defined(__gfx1250__) && __has_builtin(__builtin_amdgcn_tensor_load_to_lds)
#define HAVE_TDM 1
#else
#define HAVE_TDM 0
#endif

__device__ __forceinline__ unsigned short f2bf(float f) {
  unsigned u = __float_as_uint(f);
  u += 0x7fffu + ((u >> 16) & 1u);            // round-to-nearest-even
  return (unsigned short)(u >> 16);
}

__device__ __forceinline__ bf16x16 make_frag(uint4 lo, uint4 hi) {
  union { uint4 u[2]; bf16x16 v; } c;
  c.u[0] = lo; c.u[1] = hi;
  return c.v;
}

__device__ __forceinline__ f32x8 wmma_bf16(bf16x16 a, bf16x16 b, f32x8 c) {
#if defined(__gfx1250__)
  return __builtin_amdgcn_wmma_f32_16x16x32_bf16(false, a, false, b,
                                                 (short)0, c, false, false);
#else
  return c;
#endif
}

__device__ __forceinline__ f32x8 zero8() {
  f32x8 z = {0.f, 0.f, 0.f, 0.f, 0.f, 0.f, 0.f, 0.f};
  return z;
}

// Tensor DMA: 2D tile (tile_d0 x tile_d1 elements, 2B/elem) from a row-major
// tensor (leading-dim stride0 elements) into LDS at lds_addr. D# per ISA
// §8.3/§8.4: count=1, type=2("image"), data_size=1(2B), no pad/multicast.
__device__ __forceinline__ void tdm_load_2d(unsigned lds_addr, const void* gptr,
                                            unsigned tensor_d0,
                                            unsigned tensor_d1,
                                            unsigned tile_d0, unsigned tile_d1,
                                            unsigned stride0) {
#if HAVE_TDM
  unsigned long long ga = (unsigned long long)(uintptr_t)gptr;
  u32x4 g0;
  g0[0] = 1u;                                        // count=1, user D#
  g0[1] = lds_addr;                                  // LDS byte address
  g0[2] = (unsigned)(ga & 0xffffffffu);              // global_addr[31:0]
  g0[3] = (unsigned)((ga >> 32) & 0x01ffffffu) | (2u << 30);  // [56:32]|type=2
  i32x8 g1;
  g1[0] = (int)(1u << 16);                           // data_size=2B
  g1[1] = (int)((tensor_d0 & 0xffffu) << 16);        // tensor_dim0[15:0]
  g1[2] = (int)(((tensor_d0 >> 16) & 0xffffu) |      // tensor_dim0[31:16]
                ((tensor_d1 & 0xffffu) << 16));      // tensor_dim1[15:0]
  g1[3] = (int)(((tensor_d1 >> 16) & 0xffffu) |      // tensor_dim1[31:16]
                ((tile_d0 & 0xffffu) << 16));        // tile_dim0
  g1[4] = (int)(tile_d1 & 0xffffu);                  // tile_dim1 (tile_dim2=0)
  g1[5] = (int)stride0;                              // tensor_dim0_stride lo
  g1[6] = 0;
  g1[7] = 0;
  i32x4 z4 = {0, 0, 0, 0};
#if __clang_major__ >= 23
  i32x8 z8 = {0, 0, 0, 0, 0, 0, 0, 0};
  __builtin_amdgcn_tensor_load_to_lds(g0, g1, z4, z4, z8, 0);
#else
  __builtin_amdgcn_tensor_load_to_lds(g0, g1, z4, z4, 0);
#endif
#else
  (void)lds_addr; (void)gptr; (void)tensor_d0; (void)tensor_d1;
  (void)tile_d0; (void)tile_d1; (void)stride0;
#endif
}

__device__ __forceinline__ void wait_tensorcnt0() {
#if defined(__gfx1250__)
#if __has_builtin(__builtin_amdgcn_s_wait_tensorcnt)
  __builtin_amdgcn_s_wait_tensorcnt(0);
#else
  asm volatile("s_wait_tensorcnt 0x0" ::: "memory");
#endif
#endif
}

// ---------------------------------------------------------------------------
__global__ void cvt_x_kernel(const float* __restrict__ x,
                             unsigned short* __restrict__ xb, int n) {
  for (int i = blockIdx.x * blockDim.x + threadIdx.x; i < n;
       i += gridDim.x * blockDim.x)
    xb[i] = f2bf(x[i]);
}

// Pack+transpose Wq|Wk|Wv into bf16 Wt[3*768][768] (row = out col, col = in)
__global__ void cvt_wqkv_kernel(const float* __restrict__ Wq,
                                const float* __restrict__ Wk,
                                const float* __restrict__ Wv,
                                unsigned short* __restrict__ wt) {
  const int n = 3 * D_EMB * D_EMB;
  for (int i = blockIdx.x * blockDim.x + threadIdx.x; i < n;
       i += gridDim.x * blockDim.x) {
    int sel = i / (D_EMB * D_EMB);
    int rem = i - sel * (D_EMB * D_EMB);
    int j = rem / D_EMB;
    int k = rem - j * D_EMB;
    const float* W = (sel == 0) ? Wq : ((sel == 1) ? Wk : Wv);
    wt[i] = f2bf(W[k * D_EMB + j]);
  }
}

__global__ void cvt_wo_kernel(const float* __restrict__ Wo,
                              unsigned short* __restrict__ wt) {
  const int n = D_EMB * D_EMB;
  for (int i = blockIdx.x * blockDim.x + threadIdx.x; i < n;
       i += gridDim.x * blockDim.x) {
    int j = i / D_EMB;
    int k = i - j * D_EMB;
    wt[i] = f2bf(Wo[k * D_EMB + j]);
  }
}

// biasf[h][i][j] = bias_table[rel_index[i][j]][h]
__global__ void bias_expand_kernel(const float* __restrict__ btab,
                                   const int* __restrict__ ridx,
                                   float* __restrict__ biasf) {
  const int n = N_HEADS * SEQLEN * SEQLEN;
  for (int i = blockIdx.x * blockDim.x + threadIdx.x; i < n;
       i += gridDim.x * blockDim.x) {
    int h = i / (SEQLEN * SEQLEN);
    int rem = i - h * (SEQLEN * SEQLEN);
    biasf[i] = btab[ridx[rem] * N_HEADS + h];
  }
}

// ---------------------------------------------------------------------------
// Shared GEMM mainloop body: TDM double-buffered (or cooperative fallback).
// A: [arows][768] bf16 row-major; Bt: [brows][768] bf16 row-major (N x K).
// Each block computes a 128x128 tile; 8 waves in a 4(M) x 2(N) grid; each
// wave owns 2x4 16x16 accumulator tiles.
#define GEMM_PROLOGUE()                                                       \
  const int tid = threadIdx.x;                                                \
  const int lane = tid & 31, wave = tid >> 5;                                 \
  const int half = lane >> 4, l16 = lane & 15;                                \
  const int wm = wave & 3, wn = wave >> 2;                                    \
  const int m0 = blockIdx.x * 128, n0 = blockIdx.y * 128;                     \
  f32x8 acc[2][4];                                                            \
  _Pragma("unroll") for (int mi = 0; mi < 2; ++mi)                            \
      _Pragma("unroll") for (int ni = 0; ni < 4; ++ni) acc[mi][ni] = zero8();

#define GEMM_FRAGS_AND_MMA(AsP, BsP)                                          \
  {                                                                           \
    bf16x16 af[2], bfr[4];                                                    \
    _Pragma("unroll") for (int mi = 0; mi < 2; ++mi) {                        \
      int row = wm * 32 + mi * 16 + l16;                                      \
      uint4 lo = *(const uint4*)((AsP) + row * 32 + 8 * half);                \
      uint4 hi = *(const uint4*)((AsP) + row * 32 + 16 + 8 * half);           \
      af[mi] = make_frag(lo, hi);                                             \
    }                                                                         \
    _Pragma("unroll") for (int ni = 0; ni < 4; ++ni) {                        \
      int n = wn * 64 + ni * 16 + l16;                                        \
      uint4 lo = *(const uint4*)((BsP) + n * 32 + 8 * half);                  \
      uint4 hi = *(const uint4*)((BsP) + n * 32 + 16 + 8 * half);             \
      bfr[ni] = make_frag(lo, hi);                                            \
    }                                                                         \
    _Pragma("unroll") for (int mi = 0; mi < 2; ++mi)                          \
        _Pragma("unroll") for (int ni = 0; ni < 4; ++ni)                      \
            acc[mi][ni] = wmma_bf16(af[mi], bfr[ni], acc[mi][ni]);            \
  }

#if HAVE_TDM
#define GEMM_MAINLOOP(A, Bt, arows, brows)                                    \
  __shared__ unsigned short As[2][128 * 32];                                  \
  __shared__ unsigned short Bs[2][128 * 32];                                  \
  const bool issuer = (wave == 0);                                            \
  if (issuer) {                                                               \
    tdm_load_2d((unsigned)(uintptr_t)&As[0][0], (A) + (size_t)m0 * D_EMB,     \
                D_EMB, (arows), 32, 128, D_EMB);                              \
    tdm_load_2d((unsigned)(uintptr_t)&Bs[0][0], (Bt) + (size_t)n0 * D_EMB,    \
                D_EMB, (brows), 32, 128, D_EMB);                              \
  }                                                                           \
  for (int k0 = 0; k0 < D_EMB; k0 += 32) {                                    \
    const int cur = (k0 >> 5) & 1;                                            \
    if (issuer) wait_tensorcnt0();                                            \
    __syncthreads(); /* tiles[cur] ready; tiles[cur^1] free */                \
    if (issuer && (k0 + 32 < D_EMB)) {                                        \
      tdm_load_2d((unsigned)(uintptr_t)&As[cur ^ 1][0],                       \
                  (A) + (size_t)m0 * D_EMB + (k0 + 32), D_EMB, (arows), 32,   \
                  128, D_EMB);                                                \
      tdm_load_2d((unsigned)(uintptr_t)&Bs[cur ^ 1][0],                       \
                  (Bt) + (size_t)n0 * D_EMB + (k0 + 32), D_EMB, (brows), 32,  \
                  128, D_EMB);                                                \
    }                                                                         \
    GEMM_FRAGS_AND_MMA(&As[cur][0], &Bs[cur][0])                              \
    __syncthreads(); /* all waves done with tiles[cur] before reuse */        \
  }
#else
#define GEMM_MAINLOOP(A, Bt, arows, brows)                                    \
  __shared__ unsigned short As[128 * 32];                                     \
  __shared__ unsigned short Bs[128 * 32];                                     \
  const int lrow = tid >> 1, lcol = (tid & 1) * 16;                           \
  const uint4* ga = (const uint4*)((A) + (size_t)(m0 + lrow) * D_EMB);        \
  const uint4* gb = (const uint4*)((Bt) + (size_t)(n0 + lrow) * D_EMB);       \
  for (int k0 = 0; k0 < D_EMB; k0 += 32) {                                    \
    const int gi = (k0 + lcol) >> 3;                                          \
    uint4 a0 = ga[gi], a1 = ga[gi + 1];                                       \
    uint4 b0 = gb[gi], b1 = gb[gi + 1];                                       \
    if (k0 + 32 < D_EMB) {                                                    \
      __builtin_prefetch(&ga[gi + 4], 0, 1);                                  \
      __builtin_prefetch(&gb[gi + 4], 0, 1);                                  \
    }                                                                         \
    __syncthreads();                                                          \
    uint4* sa = (uint4*)(As + lrow * 32 + lcol);                              \
    sa[0] = a0; sa[1] = a1;                                                   \
    uint4* sb = (uint4*)(Bs + lrow * 32 + lcol);                              \
    sb[0] = b0; sb[1] = b1;                                                   \
    __syncthreads();                                                          \
    GEMM_FRAGS_AND_MMA(As, Bs)                                                \
  }
#endif

// QKV projection: x[12544,768]bf16 x Wt[2304,768]bf16 -> q/k/v [B,H,S,64]bf16
__global__ __launch_bounds__(256) void gemm_qkv_kernel(
    const unsigned short* __restrict__ A, const unsigned short* __restrict__ Bt,
    const float* __restrict__ bq, const float* __restrict__ bk,
    const float* __restrict__ bv, unsigned short* __restrict__ qo,
    unsigned short* __restrict__ ko, unsigned short* __restrict__ vo) {
  GEMM_PROLOGUE()
  GEMM_MAINLOOP(A, Bt, MROWS, NQKV)
#pragma unroll
  for (int mi = 0; mi < 2; ++mi) {
#pragma unroll
    for (int ni = 0; ni < 4; ++ni) {
      int n = n0 + wn * 64 + ni * 16 + l16;
      int sel = n / D_EMB;
      int rem = n - sel * D_EMB;
      const float* bias = (sel == 0) ? bq : ((sel == 1) ? bk : bv);
      unsigned short* dst = (sel == 0) ? qo : ((sel == 1) ? ko : vo);
      float bias_v = bias[rem];
      int h = rem >> 6, p = rem & 63;
#pragma unroll
      for (int r = 0; r < 8; ++r) {        // C layout: VGPR r -> rows r, r+8
        int m = m0 + wm * 32 + mi * 16 + r + 8 * half;
        int b = m / SEQLEN, s = m - b * SEQLEN;
        dst[(((size_t)b * N_HEADS + h) * SEQLEN + s) * PD + p] =
            f2bf(acc[mi][ni][r] + bias_v);
      }
    }
  }
}

// Output projection: att[12544,768]bf16 x Wo^T[768,768]bf16 + bo -> fp32 out
__global__ __launch_bounds__(256) void gemm_out_kernel(
    const unsigned short* __restrict__ A, const unsigned short* __restrict__ Bt,
    const float* __restrict__ bo, float* __restrict__ out) {
  GEMM_PROLOGUE()
  GEMM_MAINLOOP(A, Bt, MROWS, D_EMB)
#pragma unroll
  for (int mi = 0; mi < 2; ++mi)
#pragma unroll
    for (int ni = 0; ni < 4; ++ni) {
      int n = n0 + wn * 64 + ni * 16 + l16;
      float bias_v = bo[n];
#pragma unroll
      for (int r = 0; r < 8; ++r) {
        int m = m0 + wm * 32 + mi * 16 + r + 8 * half;
        out[(size_t)m * D_EMB + n] = acc[mi][ni][r] + bias_v;
      }
    }
}

// ---------------------------------------------------------------------------
// Fused windowed attention: one workgroup per (b,h); 13 waves, 1 q-tile each.
__global__ __launch_bounds__(416) void attn_kernel(
    const unsigned short* __restrict__ qb, const unsigned short* __restrict__ kb,
    const unsigned short* __restrict__ vb, const float* __restrict__ biasf,
    unsigned short* __restrict__ att) {
  extern __shared__ unsigned short smem[];
  unsigned short* qs = smem;                 // [208][64]
  unsigned short* ks = qs + QPAD * PD;       // [224][64]
  unsigned short* vt = ks + SPAD * PD;       // [64][224]  (V transposed)
  unsigned short* ps = vt + PD * SPAD;       // [13][16][224] per-wave P

  const int bh = blockIdx.x;
  const int b = bh / N_HEADS, h = bh - b * N_HEADS;
  const unsigned short* qg = qb + (size_t)bh * SEQLEN * PD;
  const unsigned short* kg = kb + (size_t)bh * SEQLEN * PD;
  const unsigned short* vg = vb + (size_t)bh * SEQLEN * PD;

  const int tid = threadIdx.x;
  const uint4 z4 = make_uint4(0u, 0u, 0u, 0u);

  // cooperative LDS fill (zero-padded)
  for (int i = tid; i < (QPAD * PD) / 8; i += 416) {
    int e = i * 8;
    uint4 v = (e < SEQLEN * PD) ? *(const uint4*)(qg + e) : z4;
    *(uint4*)(qs + e) = v;
  }
  for (int i = tid; i < (SPAD * PD) / 8; i += 416) {
    int e = i * 8;
    uint4 v = (e < SEQLEN * PD) ? *(const uint4*)(kg + e) : z4;
    *(uint4*)(ks + e) = v;
  }
  for (int i = tid; i < PD * SPAD; i += 416) {
    int p = i / SPAD, s = i - p * SPAD;
    vt[i] = (s < SEQLEN) ? vg[(size_t)s * PD + p] : (unsigned short)0;
  }
  __syncthreads();

  const int lane = tid & 31, wave = tid >> 5;   // wave == q row-tile (0..12)
  const int half = lane >> 4, l16 = lane & 15;

  // --- scores: S = q (16x64) @ k^T (64x224) ------------------------------
  f32x8 acc[14];
#pragma unroll
  for (int t = 0; t < 14; ++t) acc[t] = zero8();

#pragma unroll
  for (int kk = 0; kk < PD; kk += 32) {
    int row = wave * 16 + l16;
    uint4 alo = *(const uint4*)(qs + row * PD + kk + 8 * half);
    uint4 ahi = *(const uint4*)(qs + row * PD + kk + 16 + 8 * half);
    bf16x16 af = make_frag(alo, ahi);
#pragma unroll
    for (int t = 0; t < 14; ++t) {
      int n = t * 16 + l16;                      // key index (B-operand lane)
      uint4 blo = *(const uint4*)(ks + n * PD + kk + 8 * half);
      uint4 bhi = *(const uint4*)(ks + n * PD + kk + 16 + 8 * half);
      acc[t] = wmma_bf16(af, make_frag(blo, bhi), acc[t]);
    }
  }

  // --- bias + masked softmax (rows live in VGPR r; cols across 16 lanes) --
  const float* bias_h = biasf + (size_t)h * SEQLEN * SEQLEN;
  float rmax[8], rsum[8];
#pragma unroll
  for (int r = 0; r < 8; ++r) rmax[r] = -3.0e38f;
#pragma unroll
  for (int t = 0; t < 14; ++t) {
#pragma unroll
    for (int r = 0; r < 8; ++r) {
      int row = wave * 16 + r + 8 * half;
      int col = t * 16 + l16;
      float v = acc[t][r] * 0.125f;              // 1/sqrt(64)
      v += (row < SEQLEN && col < SEQLEN) ? bias_h[row * SEQLEN + col]
                                          : -1.0e30f;
      acc[t][r] = v;
      rmax[r] = fmaxf(rmax[r], v);
    }
  }
#pragma unroll
  for (int r = 0; r < 8; ++r) {                  // reduce across 16-lane half
    rmax[r] = fmaxf(rmax[r], __shfl_xor(rmax[r], 8, 32));
    rmax[r] = fmaxf(rmax[r], __shfl_xor(rmax[r], 4, 32));
    rmax[r] = fmaxf(rmax[r], __shfl_xor(rmax[r], 2, 32));
    rmax[r] = fmaxf(rmax[r], __shfl_xor(rmax[r], 1, 32));
    rsum[r] = 0.f;
  }
#pragma unroll
  for (int t = 0; t < 14; ++t)
#pragma unroll
    for (int r = 0; r < 8; ++r) {
      float e = __expf(acc[t][r] - rmax[r]);
      acc[t][r] = e;
      rsum[r] += e;
    }
#pragma unroll
  for (int r = 0; r < 8; ++r) {
    rsum[r] += __shfl_xor(rsum[r], 8, 32);
    rsum[r] += __shfl_xor(rsum[r], 4, 32);
    rsum[r] += __shfl_xor(rsum[r], 2, 32);
    rsum[r] += __shfl_xor(rsum[r], 1, 32);
    rsum[r] = 1.0f / rsum[r];
  }

  // stage P as bf16 in this wave's LDS slab (A-operand for P @ V)
  unsigned short* psw = ps + wave * 16 * SPAD;
#pragma unroll
  for (int t = 0; t < 14; ++t)
#pragma unroll
    for (int r = 0; r < 8; ++r) {
      int rl = r + 8 * half;
      psw[rl * SPAD + t * 16 + l16] = f2bf(acc[t][r] * rsum[r]);
    }
  __syncthreads();

  // --- out = P (16x224) @ V (224x64) -------------------------------------
  f32x8 o[4];
#pragma unroll
  for (int ni = 0; ni < 4; ++ni) o[ni] = zero8();
#pragma unroll
  for (int kt = 0; kt < 7; ++kt) {
    int kk = kt * 32;
    uint4 alo = *(const uint4*)(psw + l16 * SPAD + kk + 8 * half);
    uint4 ahi = *(const uint4*)(psw + l16 * SPAD + kk + 16 + 8 * half);
    bf16x16 af = make_frag(alo, ahi);
#pragma unroll
    for (int ni = 0; ni < 4; ++ni) {
      int n = ni * 16 + l16;                     // head-dim column
      uint4 blo = *(const uint4*)(vt + n * SPAD + kk + 8 * half);
      uint4 bhi = *(const uint4*)(vt + n * SPAD + kk + 16 + 8 * half);
      o[ni] = wmma_bf16(af, make_frag(blo, bhi), o[ni]);
    }
  }

  // write attention output [B,S,H*64] bf16 (input of the out-projection GEMM)
#pragma unroll
  for (int ni = 0; ni < 4; ++ni)
#pragma unroll
    for (int r = 0; r < 8; ++r) {
      int row = wave * 16 + r + 8 * half;
      if (row < SEQLEN) {
        int p = ni * 16 + l16;
        att[((size_t)b * SEQLEN + row) * D_EMB + h * PD + p] = f2bf(o[ni][r]);
      }
    }
}

// ---------------------------------------------------------------------------
extern "C" void kernel_launch(void* const* d_in, const int* in_sizes, int n_in,
                              void* d_out, int out_size, void* d_ws,
                              size_t ws_size, hipStream_t stream) {
  const float* x    = (const float*)d_in[0];
  const float* Wq   = (const float*)d_in[1];
  const float* bq   = (const float*)d_in[2];
  const float* Wk   = (const float*)d_in[3];
  const float* bk   = (const float*)d_in[4];
  const float* Wv   = (const float*)d_in[5];
  const float* bv   = (const float*)d_in[6];
  const float* Wo   = (const float*)d_in[7];
  const float* bo   = (const float*)d_in[8];
  const float* btab = (const float*)d_in[9];
  const int*   ridx = (const int*)d_in[10];
  float* out = (float*)d_out;

  char* ws = (char*)d_ws;
  size_t off = 0;
  auto carve = [&](size_t bytes) {
    char* p = ws + off;
    off = (off + bytes + 255) & ~(size_t)255;
    return p;
  };
  const size_t hb = (size_t)NBATCH * N_HEADS * SEQLEN * PD * 2;  // 19.3 MB
  unsigned short* xb   = (unsigned short*)carve((size_t)MROWS * D_EMB * 2);
  unsigned short* wtq  = (unsigned short*)carve((size_t)NQKV * D_EMB * 2);
  unsigned short* wto  = (unsigned short*)carve((size_t)D_EMB * D_EMB * 2);
  unsigned short* qws  = (unsigned short*)carve(hb);
  unsigned short* kws  = (unsigned short*)carve(hb);
  unsigned short* vws  = (unsigned short*)carve(hb);
  unsigned short* attb = (unsigned short*)carve((size_t)MROWS * D_EMB * 2);
  float*          bsf  = (float*)carve((size_t)N_HEADS * SEQLEN * SEQLEN * 4);
  (void)ws_size; (void)in_sizes; (void)n_in; (void)out_size;

  cvt_x_kernel<<<1024, 256, 0, stream>>>(x, xb, MROWS * D_EMB);
  cvt_wqkv_kernel<<<512, 256, 0, stream>>>(Wq, Wk, Wv, wtq);
  cvt_wo_kernel<<<256, 256, 0, stream>>>(Wo, wto);
  bias_expand_kernel<<<256, 256, 0, stream>>>(btab, ridx, bsf);

  gemm_qkv_kernel<<<dim3(MROWS / 128, NQKV / 128), 256, 0, stream>>>(
      xb, wtq, bq, bk, bv, qws, kws, vws);

  const size_t attn_lds =
      (size_t)(QPAD * PD + SPAD * PD + PD * SPAD + 13 * 16 * SPAD) * 2;
  attn_kernel<<<NBATCH * N_HEADS, 416, attn_lds, stream>>>(qws, kws, vws, bsf,
                                                           attb);

  gemm_out_kernel<<<dim3(MROWS / 128, D_EMB / 128), 256, 0, stream>>>(
      attb, wto, bo, out);
}
